// ManeAI_37031208026297
// MI455X (gfx1250) — matrix operand order repaired and checked by reading, not compiled, not run on hardware
//
#include <hip/hip_runtime.h>
#include <hip/hip_bf16.h>
#include <math.h>

typedef __attribute__((ext_vector_type(16))) _Float16 v16h;
typedef __attribute__((ext_vector_type(8)))  float    v8f;
typedef __attribute__((ext_vector_type(4)))  float    f32x4;

constexpr int Vn = 3;
constexpr int Nn = 200000;
constexpr int Dd = 128;
constexpr int Bb = 4096;
constexpr int Kk = 10;

// Numerically stable log_sigmoid. Argument of the log is in (1,2], where
// hardware v_log_f32 is plenty accurate -> avoid the ~30-op log1pf expansion.
__device__ __forceinline__ float logsigf(float x) {
    return fminf(x, 0.0f) - __logf(1.0f + __expf(-fabsf(x)));
}

#define CVT4(r, base, f)                 \
    r[(base) + 0] = (_Float16)(f).x;     \
    r[(base) + 1] = (_Float16)(f).y;     \
    r[(base) + 2] = (_Float16)(f).z;     \
    r[(base) + 3] = (_Float16)(f).w;

// A-matrix chunk loader (16x32, M x K; contiguous k order, same as B): lane holds M, element e maps to
// K = base + e, base = c*32 + g*16.
// So load p[0..15]. u rows are reused across 5 jobs -> cacheable.
__device__ __forceinline__ v16h load_a_chunk(const float* __restrict__ p) {
    const f32x4* q = reinterpret_cast<const f32x4*>(p);
    f32x4 f0 = q[0];  // p[0..3]
    f32x4 f1 = q[1];  // p[4..7]
    f32x4 f2 = q[2];  // p[8..11]
    f32x4 f3 = q[3];  // p[12..15]
    v16h r;
    CVT4(r, 0,  f0);
    CVT4(r, 4,  f1);
    CVT4(r, 8,  f2);
    CVT4(r, 12, f3);
    return r;
}

// B-matrix chunk loader (32x16, K x N): lane%16 = N, lane group g selects K half;
// element e maps K = c*32 + g*16 + e  => 16 contiguous floats.
// pos/neg rows are touched exactly once -> non-temporal, keep L2 for reused data.
__device__ __forceinline__ v16h load_b_chunk(const float* __restrict__ p) {
    const f32x4* q = reinterpret_cast<const f32x4*>(p);
    f32x4 f0 = __builtin_nontemporal_load(q + 0);
    f32x4 f1 = __builtin_nontemporal_load(q + 1);
    f32x4 f2 = __builtin_nontemporal_load(q + 2);
    f32x4 f3 = __builtin_nontemporal_load(q + 3);
    v16h r;
    CVT4(r, 0,  f0);
    CVT4(r, 4,  f1);
    CVT4(r, 8,  f2);
    CVT4(r, 12, f3);
    return r;
}

__global__ void zero_out_kernel(float* o) {
    if (threadIdx.x == 0 && blockIdx.x == 0) o[0] = 0.0f;
}

// b_nodes[i,b] = nodes_idx[i, shuffle[i, b+count]];   b_neigh analogous.
__global__ void build_pairs_kernel(const int* __restrict__ nodes_idx,
                                   const int* __restrict__ neigh_idx,
                                   const int* __restrict__ shuffle,
                                   const int* __restrict__ count,
                                   int* __restrict__ b_nodes,
                                   int* __restrict__ b_neigh) {
    int tid = blockIdx.x * blockDim.x + threadIdx.x;
    if (tid >= Vn * Bb) return;
    int i = tid / Bb;
    int b = tid % Bb;
    int sp = (b + count[0]) & (Bb - 1);
    int sel = shuffle[i * Bb + sp];
    b_nodes[tid] = nodes_idx[i * Bb + sel];
    b_neigh[tid] = neigh_idx[i * Bb + sel];
}

// One block = one (job p, tile of 128 b's). 8 waves/block, each wave owns 16 b's.
// Per b: 11 dot products of length 128 computed with 4 chained
// v_wmma_f32_16x16x32_f16 (A = 16 u-rows of the wave's b-tile, B = this b's
// pos/neg rows in columns 0..10). Row (b - b0) of C holds the 11 dots.
__global__ __launch_bounds__(256) void sgns_wmma_kernel(
    const float* __restrict__ node_emb, const float* __restrict__ neigh_emb,
    const int* __restrict__ neg1, const int* __restrict__ neg2,
    const int* __restrict__ neg3,
    const float* __restrict__ hyp1, const float* __restrict__ hyp2,
    const int* __restrict__ b_nodes, const int* __restrict__ b_neigh,
    float* __restrict__ out)
{
    // ---- decode job ----
    const int p = blockIdx.y;   // 0..14
    int type, i, j;
    if (p < 3) {
        type = 1; i = p; j = p;
    } else {
        const int q = (p < 9) ? (p - 3) : (p - 9);
        type = (p < 9) ? 2 : 3;
        j = q >> 1;
        const int i0 = (j == 0) ? 1 : 0;
        const int i1 = (j == 2) ? 1 : 2;
        i = (q & 1) ? i1 : i0;
    }

    const size_t tabstride = (size_t)Nn * Dd;
    const float* u_tab = node_emb + (size_t)i * tabstride;
    const int*   u_idx = b_nodes + i * Bb;

    const float* o_tab;       // table for pos AND neg rows (same in every job)
    const int*   pos_idx;
    const int*   neg_idx;
    float        wgt;
    if (type == 1) {
        o_tab   = neigh_emb + (size_t)i * tabstride;
        pos_idx = b_neigh + i * Bb;
        neg_idx = neg1 + (size_t)i * Bb * Kk;
        wgt     = -1.0f / (3.0f * Vn * Bb);
    } else if (type == 2) {
        o_tab   = node_emb + (size_t)j * tabstride;
        pos_idx = b_nodes + i * Bb;
        neg_idx = neg2 + (size_t)(j * Vn + i) * Bb * Kk;
        wgt     = -hyp1[0] / (3.0f * Vn * (Vn - 1) * Bb);
    } else {
        o_tab   = neigh_emb + (size_t)j * tabstride;
        pos_idx = b_neigh + i * Bb;
        neg_idx = neg3 + (size_t)(j * Vn + i) * Bb * Kk;
        wgt     = -hyp2[0] / (3.0f * Vn * (Vn - 1) * Bb);
    }

    const int lane = threadIdx.x & 31;
    const int wv   = threadIdx.x >> 5;              // wave in block: 0..7
    const int b0   = blockIdx.x * 128 + wv * 16;    // this wave's 16-b tile
    const int g    = lane >> 4;                     // lane half
    const int nl   = lane & 15;                     // M (for A) / N (for B/C)

    // ---- build A: 16 u rows (f32 -> f16), 4 chunks of K=32 ----
    const float* urow = u_tab + (size_t)u_idx[b0 + nl] * Dd;
    const v16h A0 = load_a_chunk(urow + 0 * 32 + g * 16);
    const v16h A1 = load_a_chunk(urow + 1 * 32 + g * 16);
    const v16h A2 = load_a_chunk(urow + 2 * 32 + g * 16);
    const v16h A3 = load_a_chunk(urow + 3 * 32 + g * 16);

    float lacc = 0.0f;

#pragma unroll
    for (int m = 0; m < 16; ++m) {
        const int b = b0 + m;
        // Column nl of B: n==0 -> positive row, 1..10 -> negatives.
        // Columns 11..15 clamp to the k=9 row: identical addresses coalesce,
        // so they add no HBM traffic and their C columns are never read.
        int kk = nl - 1;
        kk = kk < 0 ? 0 : (kk > Kk - 1 ? Kk - 1 : kk);
        const size_t ridx = (nl == 0) ? (size_t)pos_idx[b]
                                      : (size_t)neg_idx[(size_t)b * Kk + kk];
        const float* rp = o_tab + ridx * Dd;

        v8f acc = {};
        {
            v16h Bv;
            Bv  = load_b_chunk(rp + 0 * 32 + g * 16);
            acc = __builtin_amdgcn_wmma_f32_16x16x32_f16(false, A0, false, Bv,
                                                         (short)0, acc, false, false);
            Bv  = load_b_chunk(rp + 1 * 32 + g * 16);
            acc = __builtin_amdgcn_wmma_f32_16x16x32_f16(false, A1, false, Bv,
                                                         (short)0, acc, false, false);
            Bv  = load_b_chunk(rp + 2 * 32 + g * 16);
            acc = __builtin_amdgcn_wmma_f32_16x16x32_f16(false, A2, false, Bv,
                                                         (short)0, acc, false, false);
            Bv  = load_b_chunk(rp + 3 * 32 + g * 16);
            acc = __builtin_amdgcn_wmma_f32_16x16x32_f16(false, A3, false, Bv,
                                                         (short)0, acc, false, false);
        }

        // Row m of C lives in lane-half (m>=8), VGPR m&7; column = nl.
        const float x = acc[m & 7];
        if ((g == (m >> 3)) && (nl < 11)) {
            const float s = (nl == 0) ? x : -x;   // pos: logsig(+dot), neg: logsig(-dot)
            lacc += logsigf(s);
        }
    }

    // wave32 tree reduction, one atomic per wave
#pragma unroll
    for (int off = 16; off > 0; off >>= 1)
        lacc += __shfl_xor(lacc, off, 32);
    if (lane == 0)
        atomicAdd(out, lacc * wgt);
}

extern "C" void kernel_launch(void* const* d_in, const int* in_sizes, int n_in,
                              void* d_out, int out_size, void* d_ws, size_t ws_size,
                              hipStream_t stream) {
    const float* node_emb  = (const float*)d_in[0];
    const float* neigh_emb = (const float*)d_in[1];
    const int*   nodes_idx = (const int*)d_in[2];
    const int*   neigh_idx = (const int*)d_in[3];
    const int*   shuffle   = (const int*)d_in[4];
    const int*   neg1      = (const int*)d_in[5];
    const int*   neg2      = (const int*)d_in[6];
    const int*   neg3      = (const int*)d_in[7];
    const float* hyp1      = (const float*)d_in[8];
    const float* hyp2      = (const float*)d_in[9];
    const int*   count     = (const int*)d_in[10];
    float*       out       = (float*)d_out;

    int* b_nodes = (int*)d_ws;
    int* b_neigh = b_nodes + Vn * Bb;

    zero_out_kernel<<<1, 64, 0, stream>>>(out);
    build_pairs_kernel<<<(Vn * Bb + 255) / 256, 256, 0, stream>>>(
        nodes_idx, neigh_idx, shuffle, count, b_nodes, b_neigh);

    dim3 grid(Bb / 128, 15, 1);  // 32 b-tiles x 15 jobs = 480 blocks
    sgns_wmma_kernel<<<grid, 256, 0, stream>>>(
        node_emb, neigh_emb, neg1, neg2, neg3, hyp1, hyp2,
        b_nodes, b_neigh, out);
}
